// AXSLinearMixedPrecision_56126632624575
// MI455X (gfx1250) — matrix operand order repaired and checked by reading, not compiled, not run on hardware
//
#include <hip/hip_runtime.h>

typedef __attribute__((ext_vector_type(16))) __bf16 bf16x16;
typedef __attribute__((ext_vector_type(8)))  __bf16 bf16x8;
typedef __attribute__((ext_vector_type(8)))  float  f32x8;

__device__ __forceinline__ unsigned short f32_to_bf16_bits(float f) {
  unsigned int u = __float_as_uint(f);
  u += 0x7FFFu + ((u >> 16) & 1u);   // round-to-nearest-even
  return (unsigned short)(u >> 16);
}
__device__ __forceinline__ float bf16_bits_to_f32(unsigned short b) {
  return __uint_as_float(((unsigned int)b) << 16);
}
__device__ __forceinline__ float bf16_round_f32(float f) {
  return bf16_bits_to_f32(f32_to_bf16_bits(f));
}

// ---------------------------------------------------------------------------
// Blockwise absmax fake-quant (block=32, QMAX=15), fp32 in -> bf16 bits out.
// One float4 per lane; a 32-elem block spans 8 consecutive lanes.
// ---------------------------------------------------------------------------
__global__ __launch_bounds__(256) void fq_kernel(const float* __restrict__ in,
                                                 unsigned short* __restrict__ outq,
                                                 long long n4) {
  long long t = (long long)blockIdx.x * 256 + threadIdx.x;
  if (t >= n4) return;
  float4 v = reinterpret_cast<const float4*>(in)[t];
  // reference casts to bf16 first, then quantizes in fp32
  float a0 = bf16_round_f32(v.x);
  float a1 = bf16_round_f32(v.y);
  float a2 = bf16_round_f32(v.z);
  float a3 = bf16_round_f32(v.w);
  float am = fmaxf(fmaxf(fabsf(a0), fabsf(a1)), fmaxf(fabsf(a2), fabsf(a3)));
  am = fmaxf(am, __shfl_xor(am, 1, 32));
  am = fmaxf(am, __shfl_xor(am, 2, 32));
  am = fmaxf(am, __shfl_xor(am, 4, 32));
  float scale = (am > 0.0f) ? (am * (1.0f / 15.0f)) : 1.0f;
  float inv = 1.0f / scale;
  float q0 = fminf(fmaxf(rintf(a0 * inv), -15.0f), 15.0f) * scale;
  float q1 = fminf(fmaxf(rintf(a1 * inv), -15.0f), 15.0f) * scale;
  float q2 = fminf(fmaxf(rintf(a2 * inv), -15.0f), 15.0f) * scale;
  float q3 = fminf(fmaxf(rintf(a3 * inv), -15.0f), 15.0f) * scale;
  ushort4 o;
  o.x = f32_to_bf16_bits(q0);
  o.y = f32_to_bf16_bits(q1);
  o.z = f32_to_bf16_bits(q2);
  o.w = f32_to_bf16_bits(q3);
  reinterpret_cast<ushort4*>(outq)[t] = o;
}

// ---------------------------------------------------------------------------
// bf16 WMMA GEMM: C[M,N] = A[M,K] * B[N,K]^T + bias, bf16 out, fp32 accum.
// Block = 256 threads = 8 wave32, macro tile 128x256, KT=64, double-buffered
// LDS (stride 72 elems = 144B, 16B-aligned & bank-conflict-free for frag reads)
// ---------------------------------------------------------------------------
#define BM 128
#define BN 256
#define KT 64
#define LDA 72
#define LDB 72

__global__ __launch_bounds__(256) void bf16_gemm_wmma(
    const unsigned short* __restrict__ A,   // M x K  bf16 bits
    const unsigned short* __restrict__ B,   // N x K  bf16 bits
    const float* __restrict__ bias,         // N      fp32
    unsigned short* __restrict__ C,         // M x N  bf16 bits
    int M, int N, int K) {
  extern __shared__ unsigned short smem[];
  unsigned short* sA = smem;                         // 2 stages of BM*LDA
  unsigned short* sB = smem + 2 * BM * LDA;          // 2 stages of BN*LDB

  const int tid   = threadIdx.x;
  const int lane  = tid & 31;
  const int wid   = tid >> 5;
  const int half  = lane >> 4;    // 0: lanes 0-15, 1: lanes 16-31
  const int lm    = lane & 15;
  const int waveM = wid & 1;      // 2 wave rows  -> 64 M each
  const int waveN = wid >> 1;     // 4 wave cols  -> 64 N each

  const int m0 = blockIdx.y * BM;
  const int n0 = blockIdx.x * BN;

  f32x8 acc[4][4];
  {
    f32x8 z = {0.f, 0.f, 0.f, 0.f, 0.f, 0.f, 0.f, 0.f};
#pragma unroll
    for (int mi = 0; mi < 4; ++mi)
#pragma unroll
      for (int ni = 0; ni < 4; ++ni) acc[mi][ni] = z;
  }

  const int NT = K / KT;
  uint4 ra[4], rb[8];

  auto loadA = [&](int kt) {
#pragma unroll
    for (int i = 0; i < 4; ++i) {
      int c = tid + i * 256;                 // 1024 16B chunks: 8 per row
      int row = c >> 3, kc = (c & 7) << 3;
      ra[i] = *reinterpret_cast<const uint4*>(A + (size_t)(m0 + row) * K + kt * KT + kc);
    }
  };
  auto loadB = [&](int kt) {
#pragma unroll
    for (int i = 0; i < 8; ++i) {
      int c = tid + i * 256;                 // 2048 16B chunks: 8 per row
      int row = c >> 3, kc = (c & 7) << 3;
      rb[i] = *reinterpret_cast<const uint4*>(B + (size_t)(n0 + row) * K + kt * KT + kc);
    }
  };
  auto storeA = [&](int s) {
    unsigned short* d = sA + s * BM * LDA;
#pragma unroll
    for (int i = 0; i < 4; ++i) {
      int c = tid + i * 256;
      int row = c >> 3, kc = (c & 7) << 3;
      *reinterpret_cast<uint4*>(d + row * LDA + kc) = ra[i];
    }
  };
  auto storeB = [&](int s) {
    unsigned short* d = sB + s * BN * LDB;
#pragma unroll
    for (int i = 0; i < 8; ++i) {
      int c = tid + i * 256;
      int row = c >> 3, kc = (c & 7) << 3;
      *reinterpret_cast<uint4*>(d + row * LDB + kc) = rb[i];
    }
  };
  auto compute = [&](int s) {
    const unsigned short* sAb = sA + s * BM * LDA;
    const unsigned short* sBb = sB + s * BN * LDB;
#pragma unroll
    for (int ks = 0; ks < KT; ks += 32) {
      bf16x16 af[4], bfr[4];
#pragma unroll
      for (int mi = 0; mi < 4; ++mi) {
        // 16-bit A 16x32 layout: lanes 0-15 -> K0..7 & K16..23; lanes 16-31 -> K8..15 & K24..31
        int row = waveM * 64 + mi * 16 + lm;
        int kf  = ks + (half ? 8 : 0);
        bf16x8 lo = *reinterpret_cast<const bf16x8*>(sAb + row * LDA + kf);
        bf16x8 hi = *reinterpret_cast<const bf16x8*>(sAb + row * LDA + kf + 16);
        af[mi] = __builtin_shufflevector(lo, hi, 0, 1, 2, 3, 4, 5, 6, 7, 8, 9, 10, 11, 12, 13, 14, 15);
      }
#pragma unroll
      for (int ni = 0; ni < 4; ++ni) {
        // B 32x16: lane holds column N=lm, K0..7 & K16..23 (lanes 0-15) / K8..15 & K24..31 (lanes 16-31)
        int col = waveN * 64 + ni * 16 + lm;
        int kf  = ks + (half ? 8 : 0);
        bf16x8 lo = *reinterpret_cast<const bf16x8*>(sBb + col * LDB + kf);
        bf16x8 hi = *reinterpret_cast<const bf16x8*>(sBb + col * LDB + kf + 16);
        bfr[ni] = __builtin_shufflevector(lo, hi, 0, 1, 2, 3, 4, 5, 6, 7, 8, 9, 10, 11, 12, 13, 14, 15);
      }
#pragma unroll
      for (int mi = 0; mi < 4; ++mi)
#pragma unroll
        for (int ni = 0; ni < 4; ++ni)
          acc[mi][ni] = __builtin_amdgcn_wmma_f32_16x16x32_bf16(
              false, af[mi], false, bfr[ni], (short)0, acc[mi][ni], false, false);
    }
  };

  // prologue
  loadA(0);
  loadB(0);
  storeA(0);
  storeB(0);

  for (int t = 0; t < NT; ++t) {
    __syncthreads();                      // stage t&1 is ready
    if (t + 1 < NT) { loadA(t + 1); loadB(t + 1); }
    if (t + 2 < NT) {                     // warm L2 for the tile after next
      __builtin_prefetch(A + (size_t)(m0 + (tid >> 3)) * K + (t + 2) * KT + ((tid & 7) << 3), 0, 1);
      __builtin_prefetch(B + (size_t)(n0 + (tid >> 3)) * K + (t + 2) * KT + ((tid & 7) << 3), 0, 1);
    }
    compute(t & 1);
    __syncthreads();                      // all reads of the other stage done
    if (t + 1 < NT) { storeA((t + 1) & 1); storeB((t + 1) & 1); }
  }

  // epilogue: C frag layout (8 VGPRs): lanes 0-15 M=r, lanes 16-31 M=r+8, N=lm
#pragma unroll
  for (int ni = 0; ni < 4; ++ni) {
    int col = n0 + waveN * 64 + ni * 16 + lm;
    float bb = bf16_round_f32(bias[col]);   // reference casts bias to bf16
#pragma unroll
    for (int mi = 0; mi < 4; ++mi) {
      int mbase = m0 + waveM * 64 + mi * 16 + half * 8;
#pragma unroll
      for (int r = 0; r < 8; ++r) {
        float v = acc[mi][ni][r] + bb;
        C[(size_t)(mbase + r) * N + col] = f32_to_bf16_bits(v);
      }
    }
  }
}

extern "C" void kernel_launch(void* const* d_in, const int* in_sizes, int n_in,
                              void* d_out, int out_size, void* d_ws, size_t ws_size,
                              hipStream_t stream) {
  const float* x    = (const float*)d_in[0];   // (8, 2048, 4096) fp32
  const float* w    = (const float*)d_in[1];   // (16384, 4096)   fp32
  const float* bias = (const float*)d_in[2];   // (16384,)        fp32
  unsigned short* out = (unsigned short*)d_out;

  long long nx = in_sizes[0];
  long long nw = in_sizes[1];
  int N = in_sizes[2];
  int K = (int)(nw / N);
  int M = (int)(nx / K);

  // workspace: quantized bf16 copies of x and weight
  unsigned short* xq = (unsigned short*)d_ws;
  unsigned short* wq = xq + nx;

  long long nx4 = nx >> 2, nw4 = nw >> 2;
  fq_kernel<<<dim3((unsigned)((nx4 + 255) / 256)), dim3(256), 0, stream>>>(x, xq, nx4);
  fq_kernel<<<dim3((unsigned)((nw4 + 255) / 256)), dim3(256), 0, stream>>>(w, wq, nw4);

  size_t smem = (size_t)2 * (BM * LDA + BN * LDB) * sizeof(unsigned short);  // 108 KB
  static bool attr_set = false;
  (void)attr_set;
  hipFuncSetAttribute((const void*)bf16_gemm_wmma,
                      hipFuncAttributeMaxDynamicSharedMemorySize, (int)smem);

  dim3 grid((unsigned)(N / BN), (unsigned)(M / BM));
  bf16_gemm_wmma<<<grid, dim3(256), smem, stream>>>(xq, wq, bias, out, M, N, K);
}
